// ExtBlock_18554258719389
// MI455X (gfx1250) — matrix operand order repaired and checked by reading, not compiled, not run on hardware
//
#include <hip/hip_runtime.h>
#include <math.h>

// ---------------------------------------------------------------------------
// Types for CDNA5 WMMA (wave32, 16x16x32 bf16 -> f32 accumulate)
// ---------------------------------------------------------------------------
typedef __bf16 bf16_t;
typedef __attribute__((ext_vector_type(16))) __bf16 v16bf;
typedef __attribute__((ext_vector_type(8)))  float  v8f;
typedef __attribute__((ext_vector_type(4)))  unsigned int u32x4;  // 16B chunk
typedef __attribute__((ext_vector_type(4)))  unsigned int ui32x4;
typedef __attribute__((ext_vector_type(8)))  int i32x8;
typedef __attribute__((ext_vector_type(4)))  int i32x4;

#define T_TOK 7776   // B*F = 32*243
#define XDIM  1088
#define NSEED 128
#define HEADS 8
#define FDIM  243
#define BDIM  32
#define WPB   4      // waves (M tiles) per block

struct GInfo { int nj; int joints[5]; };

__device__ __constant__ int c_inv[17] = {6,0,1,2,3,4,5,7,8,9,10,11,12,13,14,15,16};

__device__ inline float gelu_f(float x) {
  return 0.5f * x * (1.0f + erff(x * 0.70710678118654752f));
}

__device__ inline float warp_sum(float v) {
#pragma unroll
  for (int o = 16; o > 0; o >>= 1) v += __shfl_xor(v, o, 32);
  return v;
}

__device__ inline float block_sum(float v, float* sm) {
  __syncthreads();
  int lane = threadIdx.x & 31, w = threadIdx.x >> 5;
  v = warp_sum(v);
  if (lane == 0) sm[w] = v;
  __syncthreads();
  int nw = (blockDim.x + 31) >> 5;
  float r = (threadIdx.x < (unsigned)nw) ? sm[threadIdx.x] : 0.0f;
  r = warp_sum(r);
  if (threadIdx.x == 0) sm[0] = r;
  __syncthreads();
  return sm[0];
}

// ---------------------------------------------------------------------------
// TDM: one tensor_load_to_lds moving a 16-row x K-col bf16 panel from global
// (row stride = ld elems) into LDS at padded row stride Kp elems, using the
// descriptor's iterate mode (16 iterations of a 1 x K tile).
// D# packing per CDNA5 ISA ch.8 (group0 128b, group1 256b, group2 128b).
// ---------------------------------------------------------------------------
__device__ inline void tdm_load_panel(const bf16_t* gsrc, long ld_elems,
                                      int K, int Kp, bf16_t* lds_dst) {
  unsigned long long ga = (unsigned long long)gsrc;
  unsigned lds = (unsigned)(unsigned long long)lds_dst;  // LDS aperture low bits
  ui32x4 g0;
  g0[0] = 1u;                                  // count=1, user descriptor
  g0[1] = lds;                                 // lds_addr [63:32]
  g0[2] = (unsigned)(ga & 0xFFFFFFFFu);        // global_addr low
  g0[3] = (unsigned)((ga >> 32) & 0x1FFFFFFu)  // global_addr[56:32]
          | (2u << 30);                        // type=2 ("image")
  i32x8 g1;
  g1[0] = (1 << 16) | (1 << 19);               // data_size=2B, iterate_enable
  g1[1] = (int)((K & 0xFFFF) << 16);           // tensor_dim0[15:0] @ bits 63:48
  g1[2] = (int)((K >> 16) & 0xFFFF)            // tensor_dim0[31:16]
          | (1 << 16);                         // tensor_dim1=1 @ bits 111:80
  g1[3] = (int)((K & 0xFFFF) << 16);           // tile_dim0=K @ bits 127:112
  g1[4] = 1;                                   // tile_dim1=1 (tile_dim2=0)
  g1[5] = (int)(ld_elems & 0xFFFFFFFF);        // tensor_dim0_stride low 32
  g1[6] = 0;                                   // stride high / dim1_stride low
  g1[7] = 0;
  i32x4 g2;
  g2[0] = 0;                                   // tensor_dim2
  g2[1] = Kp;                                  // lds_addr_increment (iterate)
  g2[2] = (int)(ld_elems & 0xFFFFFFFF);        // global_addr_increment low 32
  g2[3] = (15 << 16);                          // iterate_count=15 @ bits 127:112
  i32x4 g3 = {0, 0, 0, 0};
  i32x8 g4 = {0, 0, 0, 0, 0, 0, 0, 0};
  __builtin_amdgcn_tensor_load_to_lds(g0, g1, g2, g3, g4, 0);
}

// ---------------------------------------------------------------------------
// WMMA GEMM: C[z](M,N) = act(A[z](M,K) * Bt[z](N,K)^T + bias) (+res)
// A row-major bf16 (M,K). B given TRANSPOSED: row-major (N,K). K <= 640.
// Whole K-panels are TDM-loaded into LDS once (1 B panel shared by 4 waves +
// 1 A panel per wave); the K-loop is then pure LDS->WMMA with no barriers.
// Requirements: N % 16 == 0, K % 8 == 0, leading dims multiples of 8 elems.
// ---------------------------------------------------------------------------
struct GemmArgs {
  const bf16_t* A; long lda, a_zs;
  const bf16_t* B; long ldb, b_zs;   // transposed (N,K)
  const float* bias; long bias_zs;
  const float* res;  long ldr, r_zs;
  float*  Cf; long ldcf, cf_zs;
  bf16_t* Cb; long ldcb, cb_zs;
  int M, N, K;
  int act;   // 1: exact GELU after bias
};

union FragU { u32x4 u[2]; v16bf v; };

__global__ __launch_bounds__(128) void k_gemm(GemmArgs g) {
  extern __shared__ bf16_t smem[];   // [ B panel | A panel w0..w3 ], 16*Kp each
  int tid = threadIdx.x;
  int lane = tid & 31, w = tid >> 5;
  int n0 = blockIdx.x * 16;
  int m0 = (blockIdx.y * WPB + w) * 16;
  int z = blockIdx.z;
  bool mok = (m0 < g.M);
  const bf16_t* A  = g.A + (long)z * g.a_zs;
  const bf16_t* Bt = g.B + (long)z * g.b_zs;
  int Kp = ((g.K + 31) & ~31) + 8;   // padded LDS row stride (elems)
  bf16_t* Bp = smem;
  bf16_t* Ap = smem + (long)16 * Kp * (1 + w);
  int lrow = lane & 15, kh = lane >> 4;

  // Zero the K-tail columns [K, kpad32) of all 5 panels so the final WMMA
  // k-tile multiplies against exact zeros (only when K % 32 != 0).
  int kpad = (g.K + 31) & ~31;
  int per = kpad - g.K;
  if (per) {
    for (int i = tid; i < 5 * 16 * per; i += 128) {
      int pnl = i / (16 * per), rem = i % (16 * per);
      int rr = rem / per, cc = rem % per;
      smem[(long)pnl * 16 * Kp + (long)rr * Kp + g.K + cc] = (bf16_t)0.f;
    }
  }

  // One TDM per wave for its A panel; wave 0 also moves the B panel.
  if (mok) tdm_load_panel(A + (long)m0 * g.lda, g.lda, g.K, Kp, Ap);
  if (w == 0) tdm_load_panel(Bt + (long)n0 * g.ldb, g.ldb, g.K, Kp, Bp);
  __builtin_amdgcn_s_wait_tensorcnt(0);
  __syncthreads();

  v8f acc = {0.f,0.f,0.f,0.f,0.f,0.f,0.f,0.f};
  if (mok) {
    const bf16_t* arow = Ap + (long)lrow * Kp;
    const bf16_t* brow = Bp + (long)lrow * Kp;
    for (int k0 = 0; k0 < g.K; k0 += 32) {
      FragU fa, fb;
      // A frag: lane(lrow,kh): elems [k0+kh*8 .. +7] and [k0+16+kh*8 .. +7]
      fa.u[0] = *(const u32x4*)(arow + k0 + kh * 8);
      fa.u[1] = *(const u32x4*)(arow + k0 + 16 + kh * 8);
      // B frag: lane(lrow,kh): Bt[lrow] elems [k0+kh*8 .. +7] and [k0+16+kh*8 .. +7]
      fb.u[0] = *(const u32x4*)(brow + k0 + kh * 8);
      fb.u[1] = *(const u32x4*)(brow + k0 + 16 + kh * 8);
      acc = __builtin_amdgcn_wmma_f32_16x16x32_bf16(
          false, fa.v, false, fb.v, (short)0, acc, false, false);
    }
#pragma unroll
    for (int e = 0; e < 8; ++e) {
      int row = m0 + kh * 8 + e;
      int col = n0 + lrow;
      float v = acc[e];
      if (g.bias) v += g.bias[(long)z * g.bias_zs + col];
      if (g.act == 1) v = gelu_f(v);
      if (g.res)  v += g.res[(long)z * g.r_zs + (long)row * g.ldr + col];
      if (g.Cf) g.Cf[(long)z * g.cf_zs + (long)row * g.ldcf + col] = v;
      if (g.Cb) g.Cb[(long)z * g.cb_zs + (long)row * g.ldcb + col] = (bf16_t)v;
    }
  }
}

// ---------------------------------------------------------------------------
// Gather group joints + LayerNorm (per token).
// ---------------------------------------------------------------------------
__global__ void k_gather_ln(const float* __restrict__ x,
                            const float* __restrict__ gamma,
                            const float* __restrict__ beta,
                            float* __restrict__ xg, bf16_t* __restrict__ xln,
                            GInfo gi, int colOff) {
  __shared__ float sm[8];
  int t = blockIdx.x;
  int dg = gi.nj * 64;
  float buf[2];
  float s1 = 0.f; int it = 0;
  for (int c = threadIdx.x; c < dg; c += blockDim.x, ++it) {
    int jl = c >> 6, sub = c & 63;
    float v = x[(long)t * XDIM + gi.joints[jl] * 64 + sub];
    buf[it] = v;
    xg[(long)t * XDIM + colOff + c] = v;
    s1 += v;
  }
  float mean = block_sum(s1, sm) / (float)dg;
  float s2 = 0.f; it = 0;
  for (int c = threadIdx.x; c < dg; c += blockDim.x, ++it) {
    float d = buf[it] - mean; s2 += d * d;
  }
  float rstd = rsqrtf(block_sum(s2, sm) / (float)dg + 1e-5f);
  it = 0;
  for (int c = threadIdx.x; c < dg; c += blockDim.x, ++it) {
    float y = (buf[it] - mean) * rstd * gamma[c] + beta[c];
    xln[(long)t * XDIM + colOff + c] = (bf16_t)y;
  }
}

__global__ void k_ln(const float* __restrict__ in, long ldin, int C,
                     const float* __restrict__ gamma, const float* __restrict__ beta,
                     bf16_t* __restrict__ out, long ldout) {
  __shared__ float sm[8];
  int t = blockIdx.x;
  float buf[6];
  float s1 = 0.f; int it = 0;
  for (int c = threadIdx.x; c < C; c += blockDim.x, ++it) {
    float v = in[(long)t * ldin + c]; buf[it] = v; s1 += v;
  }
  float mean = block_sum(s1, sm) / (float)C;
  float s2 = 0.f; it = 0;
  for (int c = threadIdx.x; c < C; c += blockDim.x, ++it) {
    float d = buf[it] - mean; s2 += d * d;
  }
  float rstd = rsqrtf(block_sum(s2, sm) / (float)C + 1e-5f);
  it = 0;
  for (int c = threadIdx.x; c < C; c += blockDim.x, ++it)
    out[(long)t * ldout + c] = (bf16_t)((buf[it] - mean) * rstd * gamma[c] + beta[c]);
}

__global__ void k_seed_gather(const float* __restrict__ seed,
                              bf16_t* __restrict__ si, GInfo gi) {
  int dg = gi.nj * 64;
  long n_tot = (long)NSEED * dg;
  for (long i = blockIdx.x * (long)blockDim.x + threadIdx.x; i < n_tot;
       i += (long)gridDim.x * blockDim.x) {
    long n = i / dg; int c = (int)(i % dg);
    int jl = c >> 6, sub = c & 63;
    si[i] = (bf16_t)seed[(n * 17 + gi.joints[jl]) * 64 + sub];
  }
}

// f32 (K,N) -> bf16 transposed (N,K); coalesced writes.
__global__ void k_cvt_t(const float* __restrict__ in, bf16_t* __restrict__ out,
                        int K, int N) {
  long tot = (long)K * N;
  for (long i = blockIdx.x * (long)blockDim.x + threadIdx.x; i < tot;
       i += (long)gridDim.x * blockDim.x) {
    long n = i / K; int k = (int)(i % K);
    out[i] = (bf16_t)in[(long)k * N + n];
  }
}

// bf16 (R,C) -> bf16 (C,R)
__global__ void k_tr_bf16(const bf16_t* __restrict__ in, bf16_t* __restrict__ out,
                          int R, int C) {
  long tot = (long)R * C;
  for (long i = blockIdx.x * (long)blockDim.x + threadIdx.x; i < tot;
       i += (long)gridDim.x * blockDim.x) {
    long c = i / R; int r = (int)(i % R);
    out[i] = in[(long)r * C + c];
  }
}

// softmax over F (243) for each (b, h, n); attn layout [h][b*F+f][n]
__global__ void k_softmax(float* __restrict__ attn) {
  int b = blockIdx.x, h = blockIdx.y;
  int n = threadIdx.x; // 128
  float* p = attn + ((long)h * T_TOK + (long)b * FDIM) * 128 + n;
  float mx = -3.0e38f;
  for (int f = 0; f < FDIM; ++f) mx = fmaxf(mx, p[(long)f * 128]);
  float s = 0.f;
  for (int f = 0; f < FDIM; ++f) s += __expf(p[(long)f * 128] - mx);
  float inv = 1.0f / s;
  for (int f = 0; f < FDIM; ++f) {
    long o = (long)f * 128;
    p[o] = __expf(p[o] - mx) * inv;
  }
}

// L1-normalize over N=128 per (h,t); bf16 out + atomic mean accumulation.
__global__ void k_l1_mean(const float* __restrict__ attn, bf16_t* __restrict__ attnb,
                          float* __restrict__ mean_out, float scale) {
  __shared__ float sm[8];
  int t = blockIdx.x, h = blockIdx.y, n = threadIdx.x; // 128 threads
  long idx = ((long)h * T_TOK + t) * 128 + n;
  float v = attn[idx];
  float s = block_sum(v, sm);
  float w = v / (1e-7f + s);
  attnb[idx] = (bf16_t)w;
  atomicAdd(&mean_out[n], w * scale);
}

// joint-mixing 17x17 Linear + GELU: u (T,17,128) -> v (T,17,128)
__global__ void k_mix(const bf16_t* __restrict__ u, const float* __restrict__ Wsep,
                      const float* __restrict__ bsep, bf16_t* __restrict__ v) {
  int t = blockIdx.x, c = threadIdx.x; // 128
  float uv[17];
#pragma unroll
  for (int j = 0; j < 17; ++j) uv[j] = (float)u[((long)t * 17 + j) * 128 + c];
#pragma unroll
  for (int j2 = 0; j2 < 17; ++j2) {
    float a = bsep[j2];
#pragma unroll
    for (int j = 0; j < 17; ++j) a += uv[j] * Wsep[j * 17 + j2];
    v[((long)t * 17 + j2) * 128 + c] = (bf16_t)gelu_f(a);
  }
}

__global__ void k_combine(const float* __restrict__ x, const float* __restrict__ xcat,
                          float* __restrict__ xnew) {
  long total = (long)T_TOK * XDIM;
  for (long i = blockIdx.x * (long)blockDim.x + threadIdx.x; i < total;
       i += (long)gridDim.x * blockDim.x) {
    long t = i / XDIM; int col = (int)(i % XDIM);
    int j = col >> 6, s = col & 63;
    xnew[i] = x[i] + xcat[t * XDIM + c_inv[j] * 64 + s];
  }
}

__global__ void k_zero(float* __restrict__ p, int n) {
  int i = blockIdx.x * blockDim.x + threadIdx.x;
  if (i < n) p[i] = 0.f;
}

// ---------------------------------------------------------------------------
// Orchestration
// ---------------------------------------------------------------------------
extern "C" void kernel_launch(void* const* d_in, const int* in_sizes, int n_in,
                              void* d_out, int out_size, void* d_ws, size_t ws_size,
                              hipStream_t stream) {
  (void)in_sizes; (void)n_in; (void)out_size; (void)ws_size;
  const float* x    = (const float*)d_in[0];
  const float* seed = (const float*)d_in[1];
  auto blk = [&](int g, int idx) -> const float* {
    return (const float*)d_in[2 + 16 * g + idx];
  };
  const float* sepW1 = (const float*)d_in[82];
  const float* sepb1 = (const float*)d_in[83];
  const float* sepWs = (const float*)d_in[84];
  const float* sepbs = (const float*)d_in[85];
  const float* sepW2 = (const float*)d_in[86];
  const float* sepb2 = (const float*)d_in[87];
  const float* normg = (const float*)d_in[88];
  const float* normb = (const float*)d_in[89];

  char* base = (char*)d_ws;
  size_t off = 0;
  auto alloc = [&](size_t bytes) -> void* {
    void* p = base + off;
    off = (off + bytes + 255) & ~(size_t)255;
    return p;
  };
  const long T = T_TOK;
  float*  xg    = (float*) alloc((size_t)T * XDIM * 4);
  bf16_t* xln   = (bf16_t*)alloc((size_t)T * XDIM * 2);
  bf16_t* qb    = (bf16_t*)alloc((size_t)T * 640 * 2);
  float*  attn  = (float*) alloc((size_t)8 * T * 128 * 4);
  bf16_t* attnb = (bf16_t*)alloc((size_t)8 * T * 128 * 2);
  bf16_t* ob    = (bf16_t*)alloc((size_t)T * 640 * 2);
  float*  x1    = (float*) alloc((size_t)T * 320 * 4);
  bf16_t* x1ln  = (bf16_t*)alloc((size_t)T * 320 * 2);
  bf16_t* h1b   = (bf16_t*)alloc((size_t)T * 640 * 2);
  float*  xcat  = (float*) alloc((size_t)T * XDIM * 4);
  float*  xnew  = (float*) alloc((size_t)T * XDIM * 4);
  bf16_t* slnb  = (bf16_t*)alloc((size_t)T * XDIM * 2);
  bf16_t* ub    = (bf16_t*)alloc((size_t)T * 17 * 128 * 2);
  bf16_t* vb    = (bf16_t*)alloc((size_t)T * 17 * 128 * 2);
  bf16_t* sib   = (bf16_t*)alloc((size_t)128 * 320 * 2);
  bf16_t* w0b   = (bf16_t*)alloc((size_t)128 * 640 * 2);
  bf16_t* w1b   = (bf16_t*)alloc((size_t)128 * 640 * 2);
  bf16_t* w1tb  = (bf16_t*)alloc((size_t)640 * 128 * 2);
  bf16_t* wtb   = (bf16_t*)alloc((size_t)640 * 320 * 2);  // Wt^T (d2,dg)
  bf16_t* w0wb  = (bf16_t*)alloc((size_t)640 * 320 * 2);  // W0^T
  bf16_t* w1wb  = (bf16_t*)alloc((size_t)640 * 320 * 2);  // W1^T
  bf16_t* wpb   = (bf16_t*)alloc((size_t)320 * 640 * 2);  // Wp^T (dg,d2)
  bf16_t* wm1b  = (bf16_t*)alloc((size_t)640 * 320 * 2);  // Wm1^T (hid,dg)
  bf16_t* wm2b  = (bf16_t*)alloc((size_t)320 * 640 * 2);  // Wm2^T (dg,hid)
  bf16_t* sw1b  = (bf16_t*)alloc((size_t)128 * 64 * 2);   // sepW1^T (128,64)
  bf16_t* sw2b  = (bf16_t*)alloc((size_t)64 * 128 * 2);   // sepW2^T (64,128)

  float* out_x    = (float*)d_out;
  float* out_mean = out_x + (size_t)T * XDIM;

  const int GN[5] = {3, 3, 5, 3, 3};
  const int GJ[5][5] = {{1,2,3,0,0},{4,5,6,0,0},{0,7,8,9,10},{11,12,13,0,0},{14,15,16,0,0}};
  int colOff[5]; { int c = 0; for (int g = 0; g < 5; ++g) { colOff[g] = c; c += GN[g] * 64; } }

  auto cvt_t = [&](const float* in, bf16_t* outp, int K, int N) {
    long n = (long)K * N;
    int blocks = (int)((n + 255) / 256); if (blocks > 2048) blocks = 2048;
    k_cvt_t<<<dim3(blocks), dim3(256), 0, stream>>>(in, outp, K, N);
  };
  auto gemm = [&](const bf16_t* A, long lda, long a_zs,
                  const bf16_t* Bt, long ldb, long b_zs,
                  const float* bias, long bias_zs,
                  const float* res, long ldr, long r_zs,
                  float* Cf, long ldcf, long cf_zs,
                  bf16_t* Cb, long ldcb, long cb_zs,
                  int M, int N, int K, int Z, int act) {
    GemmArgs ga;
    ga.A = A; ga.lda = lda; ga.a_zs = a_zs;
    ga.B = Bt; ga.ldb = ldb; ga.b_zs = b_zs;
    ga.bias = bias; ga.bias_zs = bias_zs;
    ga.res = res; ga.ldr = ldr; ga.r_zs = r_zs;
    ga.Cf = Cf; ga.ldcf = ldcf; ga.cf_zs = cf_zs;
    ga.Cb = Cb; ga.ldcb = ldcb; ga.cb_zs = cb_zs;
    ga.M = M; ga.N = N; ga.K = K; ga.act = act;
    int Kp = ((K + 31) & ~31) + 8;
    size_t smem = (size_t)16 * Kp * 2 * (WPB + 1);
    dim3 grid((N + 15) / 16, (M + 16 * WPB - 1) / (16 * WPB), Z);
    k_gemm<<<grid, dim3(128), smem, stream>>>(ga);
  };

  k_zero<<<dim3(3), dim3(256), 0, stream>>>(out_mean, 5 * 128);

  for (int g = 0; g < 5; ++g) {
    GInfo gi; gi.nj = GN[g]; for (int j = 0; j < 5; ++j) gi.joints[j] = GJ[g][j];
    k_gather_ln<<<dim3((int)T), dim3(256), 0, stream>>>(
        x, blk(g, 0), blk(g, 1), xg, xln, gi, colOff[g]);
  }
  cvt_t(sepW1, sw1b, 64, 128);   // (64,128) -> (128,64)
  cvt_t(sepW2, sw2b, 128, 64);   // (128,64) -> (64,128)

  for (int g = 0; g < 5; ++g) {
    int dg = GN[g] * 64, d2 = 2 * dg, hid = 2 * dg, dh = d2 / HEADS;
    GInfo gi; gi.nj = GN[g]; for (int j = 0; j < 5; ++j) gi.joints[j] = GJ[g][j];

    k_seed_gather<<<dim3(256), dim3(256), 0, stream>>>(seed, sib, gi);
    cvt_t(blk(g, 2),  wtb,  dg, d2);    // Wt^T  (d2,dg)
    cvt_t(blk(g, 4),  w0wb, dg, d2);    // W0^T
    cvt_t(blk(g, 6),  w1wb, dg, d2);    // W1^T
    cvt_t(blk(g, 8),  wpb,  d2, dg);    // Wp^T  (dg,d2)
    cvt_t(blk(g, 12), wm1b, dg, hid);   // Wm1^T (hid,dg)
    cvt_t(blk(g, 14), wm2b, hid, dg);   // Wm2^T (dg,hid)

    // w0 = si @ W0 + b0 ; w1 = si @ W1 + b1    (128 x d2)
    gemm(sib, dg, 0, w0wb, dg, 0, blk(g, 5), 0,
         nullptr, 0, 0, nullptr, 0, 0, w0b, d2, 0, NSEED, d2, dg, 1, 0);
    gemm(sib, dg, 0, w1wb, dg, 0, blk(g, 7), 0,
         nullptr, 0, 0, nullptr, 0, 0, w1b, d2, 0, NSEED, d2, dg, 1, 0);
    k_tr_bf16<<<dim3(320), dim3(256), 0, stream>>>(w1b, w1tb, NSEED, d2);
    // q = LN1(xg) @ Wt + bt   (T x d2)
    gemm(xln + colOff[g], XDIM, 0, wtb, dg, 0, blk(g, 3), 0,
         nullptr, 0, 0, nullptr, 0, 0, qb, d2, 0, (int)T, d2, dg, 1, 0);
    // logits[h] = q_h @ w0_h^T   (T x 128, K=dh); Bt = w0b rows (N=128,K=dh)
    gemm(qb, d2, dh, w0b, d2, dh, nullptr, 0,
         nullptr, 0, 0, attn, 128, (long)T * 128, nullptr, 0, 0,
         (int)T, 128, dh, HEADS, 0);
    k_softmax<<<dim3(BDIM, HEADS), dim3(128), 0, stream>>>(attn);
    k_l1_mean<<<dim3((int)T, HEADS), dim3(128), 0, stream>>>(
        attn, attnb, out_mean + g * 128, 1.0f / (float)(BDIM * HEADS * FDIM));
    // o[h] = attn_h @ w1_h   (T x dh, K=128); Bt = w1t rows (N=dh,K=128)
    gemm(attnb, 128, (long)T * 128, w1tb, 128, (long)dh * 128, nullptr, 0,
         nullptr, 0, 0, nullptr, 0, 0, ob, d2, dh, (int)T, dh, 128, HEADS, 0);
    // x1 = xg + (o @ Wp + bp)
    gemm(ob, d2, 0, wpb, d2, 0, blk(g, 9), 0,
         xg + colOff[g], XDIM, 0, x1, dg, 0, nullptr, 0, 0,
         (int)T, dg, d2, 1, 0);
    k_ln<<<dim3((int)T), dim3(256), 0, stream>>>(
        x1, (long)dg, dg, blk(g, 10), blk(g, 11), x1ln, (long)dg);
    // MLP
    gemm(x1ln, dg, 0, wm1b, dg, 0, blk(g, 13), 0,
         nullptr, 0, 0, nullptr, 0, 0, h1b, hid, 0, (int)T, hid, dg, 1, 1);
    gemm(h1b, hid, 0, wm2b, hid, 0, blk(g, 15), 0,
         x1, dg, 0, xcat + colOff[g], XDIM, 0, nullptr, 0, 0,
         (int)T, dg, hid, 1, 0);
  }

  k_combine<<<dim3(2048), dim3(256), 0, stream>>>(x, xcat, xnew);
  k_ln<<<dim3((int)T), dim3(256), 0, stream>>>(
      xnew, (long)XDIM, XDIM, normg, normb, slnb, (long)XDIM);
  // (T,1088) viewed as (T*17,64): u = GELU(s @ W1 + b1)  (T*17 x 128)
  gemm(slnb, 64, 0, sw1b, 64, 0, sepb1, 0,
       nullptr, 0, 0, nullptr, 0, 0, ub, 128, 0, (int)(T * 17), 128, 64, 1, 1);
  k_mix<<<dim3((int)T), dim3(128), 0, stream>>>(ub, sepWs, sepbs, vb);
  // final: d_out = xnew + v @ W2 + b2
  gemm(vb, 128, 0, sw2b, 128, 0, sepb2, 0,
       xnew, 64, 0, out_x, 64, 0, nullptr, 0, 0, (int)(T * 17), 64, 128, 1, 0);
}
